// C2PSA_FMFFN_75453985457053
// MI455X (gfx1250) — matrix operand order repaired and checked by reading, not compiled, not run on hardware
//
#include <hip/hip_runtime.h>
#include <hip/hip_bf16.h>
#include <cmath>

// ---------------------------------------------------------------------------
// Types / helpers
// ---------------------------------------------------------------------------
typedef __attribute__((ext_vector_type(16))) __bf16 v16bf;
typedef __attribute__((ext_vector_type(8)))  float  v8f;
typedef __attribute__((ext_vector_type(4)))  unsigned int v4u;
typedef __attribute__((ext_vector_type(8)))  int    v8i_t;
typedef __attribute__((ext_vector_type(4)))  int    v4i_t;
typedef unsigned short u16;
typedef unsigned int   u32;

__device__ __forceinline__ u16 f2bf(float f) {
  union { float f; u32 u; } x; x.f = f;
  u32 r = x.u + 0x7FFFu + ((x.u >> 16) & 1u);   // round-to-nearest-even
  return (u16)(r >> 16);
}
__device__ __forceinline__ float bf2f(u16 h) {
  union { u32 u; float f; } x; x.u = ((u32)h) << 16; return x.f;
}

union FragAB { v16bf v; uint4 q[2]; u16 s[16]; };
union FragC  { v8f   v; float f[8]; };

#define WMMA_BF16(A, B, C) \
  __builtin_amdgcn_wmma_f32_16x16x32_bf16(false, (A), false, (B), (short)0, (C), false, false)

// Problem constants
#define MPIX   32768      // B * H * W = 32*32*32
#define CC     256        // C
#define C1     512
#define HIDN   680
#define NPAD1  688        // fc1 N padded to mult of 16
#define KPAD2  704        // fc2 K padded to mult of 32
#define KEYD   32
#define ATT_SCALE 0.17677669529663687f  // 32^-0.5

// TDM availability (probe-verified builtins; arity differs across toolchains)
#if defined(__has_builtin)
#if __has_builtin(__builtin_amdgcn_tensor_load_to_lds) && \
    __has_builtin(__builtin_amdgcn_s_wait_tensorcnt)
#define HAVE_TDM 1
#endif
#endif
#ifndef HAVE_TDM
#define HAVE_TDM 0
#endif

// ---------------------------------------------------------------------------
// Pack / convert kernels
// ---------------------------------------------------------------------------
__global__ void k_f32_to_bf16(const float* __restrict__ src, u16* __restrict__ dst, int n) {
  int i = blockIdx.x * blockDim.x + threadIdx.x;
  if (i < n) dst[i] = f2bf(src[i]);
}

// x (B, C1, HW) f32  ->  xb[(b*1024+p)*512 + c] bf16
__global__ void k_x_rows(const float* __restrict__ x, u16* __restrict__ xb) {
  int i = blockIdx.x * blockDim.x + threadIdx.x;
  if (i >= 32 * C1 * 1024) return;
  int p = i & 1023;
  int c = (i >> 10) & 511;
  int b = i >> 19;
  xb[((size_t)(b * 1024 + p)) * C1 + c] = f2bf(x[i]);
}

// fc1_w (3,680,256) -> [3][688][256] bf16, pad rows zero
__global__ void k_pack_fc1(const float* __restrict__ src, u16* __restrict__ dst) {
  int i = blockIdx.x * blockDim.x + threadIdx.x;
  if (i >= 3 * NPAD1 * CC) return;
  int c   = i & 255;
  int row = (i >> 8) % NPAD1;
  int blk = i / (NPAD1 * CC);
  dst[i] = (row < HIDN) ? f2bf(src[(size_t)blk * HIDN * CC + (size_t)row * CC + c]) : (u16)0;
}

// fc2_w (3,256,680) -> [3][256][704] bf16, pad cols zero
__global__ void k_pack_fc2(const float* __restrict__ src, u16* __restrict__ dst) {
  int i = blockIdx.x * blockDim.x + threadIdx.x;
  if (i >= 3 * CC * KPAD2) return;
  int col = i % KPAD2;
  int r   = (i / KPAD2) % CC;
  int blk = i / (CC * KPAD2);
  dst[i] = (col < HIDN) ? f2bf(src[(size_t)blk * CC * HIDN + (size_t)r * HIDN + col]) : (u16)0;
}

// zero pad columns [688,704) of t
__global__ void k_zero_pad_t(u16* __restrict__ t) {
  int i = blockIdx.x * blockDim.x + threadIdx.x;
  if (i >= MPIX * 16) return;
  int m = i >> 4;
  int c = NPAD1 + (i & 15);
  t[(size_t)m * KPAD2 + c] = 0;
}

// ---------------------------------------------------------------------------
// Generic WMMA GEMM:  Cout[M x N] = act(A[M x K] * W[N x K]^T + bias) (+C)
//   MODE 0: store bf16 row-major     MODE 1: bf16 add-to-existing
//   MODE 2: store f32 BCHW-transposed (final cv2): out[(b*512+n)*1024+p]
//   act: 0 none, 1 silu, 2 gelu(erf)
// ---------------------------------------------------------------------------
template <int MODE>
__global__ __launch_bounds__(256)
void gemm_k(const u16* __restrict__ A, int lda,
            const u16* __restrict__ W, int K,
            void* __restrict__ Cout, int ldc, int Npad,
            const float* __restrict__ bias, int Nbias, int act) {
  const int lane    = threadIdx.x & 31;
  const int wave    = threadIdx.x >> 5;
  const int mw      = wave & 3;           // 4 M-subtiles of 16 -> 64
  const int nw      = wave >> 2;          // 2 N-subtiles of 64 -> 128
  const int mBase   = blockIdx.x * 64 + mw * 16;
  const int nBase0  = blockIdx.y * 128 + nw * 64;
  const int halfSel = lane >> 4;
  const int lidx    = lane & 15;

  const v8f vz = {0.f, 0.f, 0.f, 0.f, 0.f, 0.f, 0.f, 0.f};
  v8f acc[4];
  bool ok[4];
#pragma unroll
  for (int nt = 0; nt < 4; nt++) { acc[nt] = vz; ok[nt] = (nBase0 + nt * 16) < Npad; }

  const u16* Arow = A + (size_t)(mBase + lidx) * lda;
  const int kA = halfSel * 16;
  const int kB = halfSel * 16;

  for (int ks = 0; ks < K; ks += 32) {
    FragAB a;
    a.q[0] = *(const uint4*)(Arow + ks + kA);
    a.q[1] = *(const uint4*)(Arow + ks + kA + 8);
    if (ks + 64 < K) __builtin_prefetch(Arow + ks + 64, 0, 1);  // global_prefetch_b8
#pragma unroll
    for (int nt = 0; nt < 4; nt++) {
      if (!ok[nt]) continue;
      const u16* Wrow = W + (size_t)(nBase0 + nt * 16 + lidx) * K + ks + kB;
      FragAB b;
      b.q[0] = *(const uint4*)(Wrow);
      b.q[1] = *(const uint4*)(Wrow + 8);
      acc[nt] = WMMA_BF16(a.v, b.v, acc[nt]);
    }
  }

#pragma unroll
  for (int nt = 0; nt < 4; nt++) {
    if (!ok[nt]) continue;
    int n = nBase0 + nt * 16 + lidx;
    float bv = (bias != nullptr && n < Nbias) ? bias[n] : 0.f;
    FragC c; c.v = acc[nt];
#pragma unroll
    for (int r = 0; r < 8; r++) {
      int m = mBase + r + halfSel * 8;
      float val = c.f[r] + bv;
      if (act == 1)      val = val / (1.f + __expf(-val));
      else if (act == 2) val = 0.5f * val * (1.f + erff(val * 0.7071067811865475f));
      if (MODE == 0) {
        ((u16*)Cout)[(size_t)m * ldc + n] = f2bf(val);
      } else if (MODE == 1) {
        u16* cp = (u16*)Cout + (size_t)m * ldc + n;
        *cp = f2bf(bf2f(*cp) + val);
      } else {
        int b = m >> 10, p = m & 1023;
        ((float*)Cout)[(((size_t)b * 512 + n) << 10) | p] = val;
      }
    }
  }
}

// ---------------------------------------------------------------------------
// TDM: DMA the V' tile (1024 rows x 64 bf16 cols, row stride 512 elems) into
// LDS at byte offset ldsOff.  D# per CDNA5 ISA ch.8 (2D tensor, groups 2/3
// zero/NULL-equivalent).  Issued once per workgroup, overlapped with QK^T.
// ---------------------------------------------------------------------------
#if HAVE_TDM
__device__ __forceinline__ void tdm_load_v(const u16* gsrc, unsigned ldsOff) {
  unsigned long long ga = (unsigned long long)(size_t)gsrc;
  // group0: count=1 | lds_addr | global_addr[56:0] | type=2 ("image")
  v4u g0 = { 1u,
             ldsOff,
             (u32)ga,
             ((u32)(ga >> 32) & 0x01FFFFFFu) | 0x80000000u };
  // group1: data_size=2B; tensor_dim0=64, tensor_dim1=1024;
  //         tile_dim0=64, tile_dim1=1024, tile_dim2=0; dim0_stride=512
  v8i_t g1 = { (int)0x00010000,          // data_size=1 (2 bytes), mask=0
               (int)(64u << 16),          // tensor_dim0 lo16 @ bits[63:48]
               (int)(1024u << 16),        // tensor_dim1 lo16 @ bits[95:80]
               (int)(64u << 16),          // tile_dim0 @ bits[127:112]
               (int)1024,                 // tile_dim1 (tile_dim2=0)
               (int)512,                  // tensor_dim0_stride lo32
               0, 0 };
  v4i_t z4 = {0, 0, 0, 0};
#if __clang_major__ >= 23
  v8i_t z8 = {0, 0, 0, 0, 0, 0, 0, 0};
  __builtin_amdgcn_tensor_load_to_lds(g0, g1, z4, z4, z8, 0);
#else
  __builtin_amdgcn_tensor_load_to_lds(g0, g1, z4, z4, 0);
#endif
}
#endif

// ---------------------------------------------------------------------------
// Fused attention: per (b, head, 16-row tile).
//   TDM-prefetch V' -> LDS; S = Q K^T (WMMA) -> LDS f32; softmax; P bf16 in
//   LDS; O = P V (WMMA) with B-fragments gathered from LDS.
//   Dynamic LDS: S 64KB + P 32KB + V 128KB = 224KB.
// ---------------------------------------------------------------------------
#define LDS_S_BYTES (16 * 1024 * 4)
#define LDS_P_BYTES (16 * 1024 * 2)
#define LDS_V_OFF   (LDS_S_BYTES + LDS_P_BYTES)   // 98304
#define LDS_TOTAL   (LDS_V_OFF + 1024 * 64 * 2)   // 229376

__global__ __launch_bounds__(128)
void attn_k(const u16* __restrict__ qkv, u16* __restrict__ attn_out) {
  extern __shared__ char smem[];
  float* S    = (float*)smem;                     // 16 x 1024 f32
  u16*   P    = (u16*)(smem + LDS_S_BYTES);       // 16 x 1024 bf16
  u16*   Vlds = (u16*)(smem + LDS_V_OFF);         // 1024 x 64 bf16

  const int lane    = threadIdx.x & 31;
  const int wave    = threadIdx.x >> 5;           // 0..3
  const int bh      = blockIdx.y;
  const int b       = bh >> 2, h = bh & 3;
  const int row0    = blockIdx.x * 16;
  const int halfSel = lane >> 4;
  const int lidx    = lane & 15;
  const v8f vz = {0.f, 0.f, 0.f, 0.f, 0.f, 0.f, 0.f, 0.f};

  const u16* base = qkv + ((size_t)b * 1024) * 512 + h * 128;

  // ---- Phase 0: kick off async V' staging (overlaps with QK^T) ----
#if HAVE_TDM
  if (wave == 0) tdm_load_v(base + 64, LDS_V_OFF);
#else
  for (int r = threadIdx.x; r < 1024; r += 128) {
    const uint4* src = (const uint4*)(base + (size_t)r * 512 + 64);
    uint4* dst = (uint4*)(Vlds + r * 64);
#pragma unroll
    for (int q = 0; q < 8; q++) dst[q] = src[q];
  }
#endif

  // ---- Phase 1: S = Q K^T ----
  FragAB a;
  {
    const u16* qrow = base + (size_t)(row0 + lidx) * 512 + halfSel * 16;
    a.q[0] = *(const uint4*)(qrow);
    a.q[1] = *(const uint4*)(qrow + 8);
  }
  for (int nt = 0; nt < 16; nt++) {
    int j0 = wave * 256 + nt * 16;
    const u16* krow = base + (size_t)(j0 + lidx) * 512 + KEYD + halfSel * 16;
    FragAB kb;
    kb.q[0] = *(const uint4*)(krow);
    kb.q[1] = *(const uint4*)(krow + 8);
    v8f c = vz;
    c = WMMA_BF16(a.v, kb.v, c);
    FragC cc; cc.v = c;
#pragma unroll
    for (int r = 0; r < 8; r++) {
      int m = r + halfSel * 8;
      S[m * 1024 + j0 + lidx] = cc.f[r];
    }
  }
  __syncthreads();

  // ---- Phase 2: softmax rows (scaled) ----
  for (int rr = 0; rr < 4; rr++) {
    int r = wave * 4 + rr;
    float mx = -1e30f;
    for (int j = lane; j < 1024; j += 32) mx = fmaxf(mx, S[r * 1024 + j]);
    for (int o = 16; o > 0; o >>= 1) mx = fmaxf(mx, __shfl_xor(mx, o, 32));
    float sum = 0.f;
    for (int j = lane; j < 1024; j += 32) {
      float e = __expf(ATT_SCALE * (S[r * 1024 + j] - mx));
      S[r * 1024 + j] = e;
      sum += e;
    }
    for (int o = 16; o > 0; o >>= 1) sum += __shfl_xor(sum, o, 32);
    float rn = 1.f / sum;
    for (int j = lane; j < 1024; j += 32) P[r * 1024 + j] = f2bf(S[r * 1024 + j] * rn);
  }

  // ---- ensure TDM V' tile has landed before P*V ----
#if HAVE_TDM
  if (wave == 0) __builtin_amdgcn_s_wait_tensorcnt(0);
#endif
  __syncthreads();

  // ---- Phase 3: O = P V  (each wave: one 16-wide d tile) ----
  v8f acc = vz;
  const int d = wave * 16 + lidx;
  for (int ks = 0; ks < 1024; ks += 32) {
    FragAB pa;
    const u16* prow = P + lidx * 1024 + ks + halfSel * 16;
    pa.q[0] = *(const uint4*)(prow);
    pa.q[1] = *(const uint4*)(prow + 8);
    FragAB vb;
    int jb = ks + halfSel * 16;
#pragma unroll
    for (int t = 0; t < 16; t++) vb.s[t] = Vlds[(size_t)(jb + t) * 64 + d];
    acc = WMMA_BF16(pa.v, vb.v, acc);
  }
  FragC cc; cc.v = acc;
#pragma unroll
  for (int r = 0; r < 8; r++) {
    int m = row0 + r + halfSel * 8;
    attn_out[((size_t)(b * 1024 + m)) * CC + h * 64 + d] = f2bf(cc.f[r]);
  }
}

// ---------------------------------------------------------------------------
// Depthwise 3x3 positional conv on V, added into attn_out
// ---------------------------------------------------------------------------
__global__ void pe_k(const u16* __restrict__ qkv, const float* __restrict__ pew,
                     u16* __restrict__ attn_out) {
  int idx = blockIdx.x * blockDim.x + threadIdx.x;
  if (idx >= MPIX * CC) return;
  int c   = idx & 255;
  int pix = idx >> 8;
  int b   = pix >> 10;
  int p   = pix & 1023;
  int y = p >> 5, x = p & 31;
  int qc = (c >> 6) * 128 + 64 + (c & 63);
  const u16* qb = qkv + ((size_t)b << 10) * 512;
  float acc = 0.f;
#pragma unroll
  for (int dy = 0; dy < 3; dy++) {
    int yy = y + dy - 1;
    if ((unsigned)yy >= 32u) continue;
#pragma unroll
    for (int dx = 0; dx < 3; dx++) {
      int xx = x + dx - 1;
      if ((unsigned)xx >= 32u) continue;
      acc += pew[c * 9 + dy * 3 + dx] * bf2f(qb[(size_t)(yy * 32 + xx) * 512 + qc]);
    }
  }
  attn_out[idx] = f2bf(bf2f(attn_out[idx]) + acc);
}

// ---------------------------------------------------------------------------
// 4x4 windowed rFFT2 * cw * irFFT2 (ortho), in-place on t.
// ---------------------------------------------------------------------------
__global__ void fft_k(u16* __restrict__ t, const float* __restrict__ cw) {
  const int TOT = 32 * 64 * HIDN;
  int idx = blockIdx.x * blockDim.x + threadIdx.x;
  if (idx >= TOT) return;
  int ch = idx % HIDN;
  int wi = (idx / HIDN) & 63;
  int b  = idx / (HIDN * 64);
  int wy = wi >> 3, wx = wi & 7;

  float g[4][4];
#pragma unroll
  for (int dy = 0; dy < 4; dy++)
#pragma unroll
    for (int dx = 0; dx < 4; dx++) {
      int y = wy * 4 + dy, x = wx * 4 + dx;
      g[dy][dx] = bf2f(t[((size_t)(b * 1024 + y * 32 + x)) * KPAD2 + ch]);
    }

  const float cre[4] = {1.f, 0.f, -1.f, 0.f};   // Re{(±i)^k}
  const float sfw[4] = {0.f, -1.f, 0.f, 1.f};   // Im{(-i)^k}
  const float sin_[4] = {0.f, 1.f, 0.f, -1.f};  // Im{(+i)^k}

  float Fr[4][3], Fi[4][3];
#pragma unroll
  for (int u = 0; u < 4; u++)
#pragma unroll
    for (int v = 0; v < 3; v++) {
      float sr = 0.f, si = 0.f;
#pragma unroll
      for (int y = 0; y < 4; y++)
#pragma unroll
        for (int x = 0; x < 4; x++) {
          int k = (u * y + v * x) & 3;
          sr += g[y][x] * cre[k];
          si += g[y][x] * sfw[k];
        }
      Fr[u][v] = 0.25f * sr;
      Fi[u][v] = 0.25f * si;
    }

  float Gr[4][4], Gi[4][4];
#pragma unroll
  for (int u = 0; u < 4; u++)
#pragma unroll
    for (int v = 0; v < 3; v++) {
      float wr = cw[((u * 3 + v) * HIDN + ch) * 2 + 0];
      float wiI = cw[((u * 3 + v) * HIDN + ch) * 2 + 1];
      Gr[u][v] = Fr[u][v] * wr - Fi[u][v] * wiI;
      Gi[u][v] = Fr[u][v] * wiI + Fi[u][v] * wr;
    }
#pragma unroll
  for (int u = 0; u < 4; u++) {
    int uu = (4 - u) & 3;
    Gr[u][3] = Gr[uu][1];
    Gi[u][3] = -Gi[uu][1];
  }

#pragma unroll
  for (int y = 0; y < 4; y++)
#pragma unroll
    for (int x = 0; x < 4; x++) {
      float s = 0.f;
#pragma unroll
      for (int u = 0; u < 4; u++)
#pragma unroll
        for (int v = 0; v < 4; v++) {
          int k = (u * y + v * x) & 3;
          s += Gr[u][v] * cre[k] - Gi[u][v] * sin_[k];
        }
      int yy = wy * 4 + y, xx = wx * 4 + x;
      t[((size_t)(b * 1024 + yy * 32 + xx)) * KPAD2 + ch] = f2bf(0.25f * s);
    }
}

// ---------------------------------------------------------------------------
// Host: full pipeline
// ---------------------------------------------------------------------------
extern "C" void kernel_launch(void* const* d_in, const int* in_sizes, int n_in,
                              void* d_out, int out_size, void* d_ws, size_t ws_size,
                              hipStream_t stream) {
  const float* x      = (const float*)d_in[0];
  const float* cv1_w  = (const float*)d_in[1];
  const float* cv2_w  = (const float*)d_in[2];
  const float* qkv_w  = (const float*)d_in[3];
  const float* proj_w = (const float*)d_in[4];
  const float* pe_w   = (const float*)d_in[5];
  const float* fc1_w  = (const float*)d_in[6];
  const float* fc1_b  = (const float*)d_in[7];
  const float* cw     = (const float*)d_in[8];
  const float* fc2_w  = (const float*)d_in[9];
  const float* fc2_b  = (const float*)d_in[10];

  char* ws = (char*)d_ws;
  size_t off = 0;
  auto alloc = [&](size_t bytes) -> char* {
    char* p = ws + off;
    off += (bytes + 255) & ~(size_t)255;
    return p;
  };
  u16* xb    = (u16*)alloc((size_t)MPIX * C1 * 2);      // 32 MB
  u16* ybuf  = (u16*)alloc((size_t)MPIX * C1 * 2);      // 32 MB: [:,0:256]=a, [:,256:512]=bf
  u16* qkvb  = (u16*)alloc((size_t)MPIX * C1 * 2);      // 32 MB
  u16* atto  = (u16*)alloc((size_t)MPIX * CC * 2);      // 16 MB
  u16* tbuf  = (u16*)alloc((size_t)MPIX * KPAD2 * 2);   // 44 MB
  u16* wcv1  = (u16*)alloc((size_t)C1 * C1 * 2);
  u16* wcv2  = (u16*)alloc((size_t)C1 * C1 * 2);
  u16* wqkv  = (u16*)alloc((size_t)3 * C1 * CC * 2);
  u16* wproj = (u16*)alloc((size_t)3 * CC * CC * 2);
  u16* wfc1  = (u16*)alloc((size_t)3 * NPAD1 * CC * 2);
  u16* wfc2  = (u16*)alloc((size_t)3 * CC * KPAD2 * 2);
  if (ws_size < off) return;  // workspace too small; nothing safe to do

  const int TPB = 256;
  auto cdiv = [](int a, int b) { return (a + b - 1) / b; };

  // ---- weight + input packing ----
  k_f32_to_bf16<<<cdiv(C1 * C1, TPB), TPB, 0, stream>>>(cv1_w, wcv1, C1 * C1);
  k_f32_to_bf16<<<cdiv(C1 * C1, TPB), TPB, 0, stream>>>(cv2_w, wcv2, C1 * C1);
  k_f32_to_bf16<<<cdiv(3 * C1 * CC, TPB), TPB, 0, stream>>>(qkv_w, wqkv, 3 * C1 * CC);
  k_f32_to_bf16<<<cdiv(3 * CC * CC, TPB), TPB, 0, stream>>>(proj_w, wproj, 3 * CC * CC);
  k_pack_fc1<<<cdiv(3 * NPAD1 * CC, TPB), TPB, 0, stream>>>(fc1_w, wfc1);
  k_pack_fc2<<<cdiv(3 * CC * KPAD2, TPB), TPB, 0, stream>>>(fc2_w, wfc2);
  k_x_rows<<<cdiv(32 * C1 * 1024, TPB), TPB, 0, stream>>>(x, xb);
  k_zero_pad_t<<<cdiv(MPIX * 16, TPB), TPB, 0, stream>>>(tbuf);

  // ---- cv1 + silu:  ybuf[M,512] = silu(xb * cv1^T) ----
  gemm_k<0><<<dim3(MPIX / 64, 4), 256, 0, stream>>>(xb, C1, wcv1, C1, ybuf, C1, C1,
                                                    nullptr, 0, 1);

  for (int i = 0; i < 3; i++) {
    // qkv = bf * qkv_w^T : [M,512]
    gemm_k<0><<<dim3(MPIX / 64, 4), 256, 0, stream>>>(ybuf + 256, C1,
                                                      wqkv + (size_t)i * C1 * CC, CC,
                                                      qkvb, C1, C1, nullptr, 0, 0);
    // fused attention (TDM V-stage, S, softmax, P*V) -> atto
    attn_k<<<dim3(64, 128), 128, LDS_TOTAL, stream>>>(qkvb, atto);
    // + depthwise 3x3 on V
    pe_k<<<cdiv(MPIX * CC, TPB), TPB, 0, stream>>>(qkvb, pe_w + (size_t)i * CC * 9, atto);
    // bf += atto * proj^T
    gemm_k<1><<<dim3(MPIX / 64, 2), 256, 0, stream>>>(atto, CC,
                                                      wproj + (size_t)i * CC * CC, CC,
                                                      ybuf + 256, C1, CC, nullptr, 0, 0);
    // t = gelu(bf * fc1^T + fc1_b) : [M,688] (pad cols zero), stride 704
    gemm_k<0><<<dim3(MPIX / 64, 6), 256, 0, stream>>>(ybuf + 256, C1,
                                                      wfc1 + (size_t)i * NPAD1 * CC, CC,
                                                      tbuf, KPAD2, NPAD1,
                                                      fc1_b + (size_t)i * HIDN, HIDN, 2);
    // windowed FFT filter in place
    fft_k<<<cdiv(32 * 64 * HIDN, TPB), TPB, 0, stream>>>(tbuf, cw + (size_t)i * 4 * 3 * HIDN * 2);
    // bf += t * fc2^T + fc2_b
    gemm_k<1><<<dim3(MPIX / 64, 2), 256, 0, stream>>>(tbuf, KPAD2,
                                                      wfc2 + (size_t)i * CC * KPAD2, KPAD2,
                                                      ybuf + 256, C1, CC,
                                                      fc2_b + (size_t)i * CC, CC, 0);
  }

  // ---- cv2 + silu, f32 BCHW-transposed store to d_out ----
  gemm_k<2><<<dim3(MPIX / 64, 4), 256, 0, stream>>>(ybuf, C1, wcv2, C1, d_out, 0, C1,
                                                    nullptr, 0, 1);
}
